// MyBuddyModel_25494925869686
// MI455X (gfx1250) — matrix operand order repaired and checked by reading, not compiled, not run on hardware
//
#include <hip/hip_runtime.h>
#include <hip/hip_bf16.h>

// ---------------------------------------------------------------------------
// MI455X (gfx1250) implementation of the TTS acoustic model.
// All GEMM-shaped compute runs on v_wmma_f32_16x16x32_f16. Each wave32
// computes a 32 x (16*NACC) strip (MACC=2 row tiles x NACC col tiles):
// per k-step, 2 A fragments + NACC B fragments are staged into distinct
// registers (clause-able loads, staggered s_wait_loadcnt), then 2*NACC WMMAs
// issue back-to-back. This cuts VMEM issue per WMMA to ~1.5 b128 and avoids
// the load->full-wait->wmma lockstep. GEMMs stream from the 192MB L2.
// ---------------------------------------------------------------------------

#define Bz   32
#define Ttok 512
#define DIN  256
#define Hd   192
#define NH   4
#define DH   48
#define DHP  64
#define TFr  2560
#define PADT 8
#define TPAD (TFr + 2 * PADT)   // 2576
#define MELC 80
#define DC   256

typedef _Float16 v16h __attribute__((ext_vector_type(16)));
typedef _Float16 v8h  __attribute__((ext_vector_type(8)));
typedef float    v8f  __attribute__((ext_vector_type(8)));

// ---- WMMA fragment loaders (layouts per CDNA5 ISA 7.12.2) ------------------
__device__ __forceinline__ v16h ld_a_frag(const _Float16* A, int lda, int k0, int lane) {
  const int hi = (lane >> 4) & 1, r = lane & 15;
  const _Float16* p = A + (size_t)r * lda + k0 + hi * 8;
  v8h lo = *(const v8h*)(p);
  v8h hh = *(const v8h*)(p + 16);
  v16h o;
#pragma unroll
  for (int i = 0; i < 8; ++i) { o[i] = lo[i]; o[8 + i] = hh[i]; }
  return o;
}

__device__ __forceinline__ v16h ld_b_frag(const _Float16* Bp, int ldb, int k0, int lane) {
  const int hi = (lane >> 4) & 1, r = lane & 15;
  const _Float16* p = Bp + (size_t)r * ldb + k0 + hi * 8;
  v8h lo = *(const v8h*)(p);
  v8h hh = *(const v8h*)(p + 16);
  v16h o;
#pragma unroll
  for (int i = 0; i < 8; ++i) { o[i] = lo[i]; o[8 + i] = hh[i]; }
  return o;
}

// MACC row tiles x NACC col tiles per wave; fragments staged per k-step.
// acc is flat: acc[m * NACC + u].
template <int K, int NACC, int MACC>
__device__ __forceinline__ void gemm_mm(v8f* acc, const _Float16* A, int lda,
                                        const _Float16* Bp, int ldb, int lane) {
#pragma unroll
  for (int k0 = 0; k0 < K; k0 += 32) {
    v16h a[MACC];
#pragma unroll
    for (int m = 0; m < MACC; ++m)
      a[m] = ld_a_frag(A + (size_t)(m * 16) * lda, lda, k0, lane);
    v16h b[NACC];
#pragma unroll
    for (int u = 0; u < NACC; ++u)
      b[u] = ld_b_frag(Bp + (size_t)(u * 16) * ldb, ldb, k0, lane);
#pragma unroll
    for (int u = 0; u < NACC; ++u)
#pragma unroll
      for (int m = 0; m < MACC; ++m)
        acc[m * NACC + u] = __builtin_amdgcn_wmma_f32_16x16x32_f16(
            false, a[m], false, b[u], (short)0, acc[m * NACC + u], false, false);
  }
}

// ---- Utility kernels -------------------------------------------------------
__global__ void k_zero16(uint4* p, size_t n) {
  size_t i = (size_t)blockIdx.x * blockDim.x + threadIdx.x;
  size_t st = (size_t)gridDim.x * blockDim.x;
  uint4 z = make_uint4(0u, 0u, 0u, 0u);
  for (; i < n; i += st) p[i] = z;
}

__global__ void k_f32h(const float* __restrict__ in, _Float16* __restrict__ out, int n) {
  int i = blockIdx.x * blockDim.x + threadIdx.x;
  if (i < n) out[i] = (_Float16)in[i];
}

// conv weight [O=256][I=256][3] -> f16 [s][O][I]
__global__ void k_convw(const float* __restrict__ w, _Float16* __restrict__ out) {
  int i = blockIdx.x * blockDim.x + threadIdx.x;
  if (i >= 3 * 256 * 256) return;
  int s = i / 65536, r = i & 65535, n = r >> 8, c = r & 255;
  out[i] = (_Float16)w[n * 768 + c * 3 + s];
}

// ---- Input projection: x = bert @ in_w.T + in_b + emo[ids] ----------------
// Wave strip: 32 rows x 64 cols (3 col groups cover N=192).
__global__ void k_inproj(const _Float16* __restrict__ A16, const _Float16* __restrict__ W,
                         const float* __restrict__ bias, const float* __restrict__ emo,
                         const int* __restrict__ eid, float* __restrict__ x32) {
  int wid = blockIdx.x * (blockDim.x >> 5) + (threadIdx.x >> 5);
  int lane = threadIdx.x & 31;
  if (wid >= (Bz * Ttok / 32) * 3) return;
  int ng = wid % 3, mt = wid / 3, row0 = mt * 32, n0 = ng * 64;
  v8f acc[8] = {};
  gemm_mm<DIN, 4, 2>(acc, A16 + (size_t)row0 * DIN, DIN, W + (size_t)n0 * DIN, DIN, lane);
  int hi = (lane >> 4) & 1, l16 = lane & 15;
  int b = row0 >> 9;
  const float* er = emo + eid[b] * Hd;
#pragma unroll
  for (int m = 0; m < 2; ++m)
#pragma unroll
    for (int u = 0; u < 4; ++u) {
      int n = n0 + u * 16 + l16;
      float add = bias[n] + er[n];
#pragma unroll
      for (int j = 0; j < 8; ++j)
        x32[(size_t)(row0 + m * 16 + j + hi * 8) * Hd + n] = acc[m * 4 + u][j] + add;
    }
}

// ---- LayerNorm over H=192, write f16 --------------------------------------
__global__ void k_ln(const float* __restrict__ x32, const float* __restrict__ g,
                     const float* __restrict__ bt, _Float16* __restrict__ nh) {
  __shared__ float s1[64], s2[64];
  int row = blockIdx.x, tid = threadIdx.x;
  const float* xr = x32 + (size_t)row * Hd;
  float a0 = xr[tid], a1 = xr[tid + 64], a2 = xr[tid + 128];
  s1[tid] = a0 + a1 + a2;
  s2[tid] = a0 * a0 + a1 * a1 + a2 * a2;
  __syncthreads();
#pragma unroll
  for (int st = 32; st > 0; st >>= 1) {
    if (tid < st) { s1[tid] += s1[tid + st]; s2[tid] += s2[tid + st]; }
    __syncthreads();
  }
  float mean = s1[0] * (1.f / Hd);
  float var = s2[0] * (1.f / Hd) - mean * mean;
  float rs = rsqrtf(var + 1e-5f);
  _Float16* orow = nh + (size_t)row * Hd;
  orow[tid]       = (_Float16)((a0 - mean) * rs * g[tid] + bt[tid]);
  orow[tid + 64]  = (_Float16)((a1 - mean) * rs * g[tid + 64] + bt[tid + 64]);
  orow[tid + 128] = (_Float16)((a2 - mean) * rs * g[tid + 128] + bt[tid + 128]);
}

// ---- QKV GEMM, scatter into q(+scale)/k (head-padded) and v^T -------------
// Wave strip: 32 rows x 64 cols (9 col groups cover N=576).
__global__ void k_qkv(const _Float16* __restrict__ A16, const _Float16* __restrict__ W,
                      const float* __restrict__ bias, _Float16* __restrict__ qh,
                      _Float16* __restrict__ kh, _Float16* __restrict__ vt) {
  int wid = blockIdx.x * (blockDim.x >> 5) + (threadIdx.x >> 5);
  int lane = threadIdx.x & 31;
  if (wid >= (Bz * Ttok / 32) * 9) return;
  int ng = wid % 9, mt = wid / 9, row0 = mt * 32, n0 = ng * 64;
  v8f acc[8] = {};
  gemm_mm<Hd, 4, 2>(acc, A16 + (size_t)row0 * Hd, Hd, W + (size_t)n0 * Hd, Hd, lane);
  int hi = (lane >> 4) & 1, l16 = lane & 15;
  int b = row0 >> 9;
  const float SCL = 0.14433756729740643f;  // 1/sqrt(48)
#pragma unroll
  for (int m = 0; m < 2; ++m)
#pragma unroll
    for (int u = 0; u < 4; ++u) {
      int n = n0 + u * 16 + l16;
      float bb = bias[n];
#pragma unroll
      for (int j = 0; j < 8; ++j) {
        int t = (row0 + m * 16 + j + hi * 8) & 511;
        float v = acc[m * 4 + u][j] + bb;
        if (n < Hd) {
          int h = n / DH, d = n - h * DH;
          qh[((size_t)((b * NH + h) * Ttok) + t) * DHP + d] = (_Float16)(v * SCL);
        } else if (n < 2 * Hd) {
          int nn = n - Hd; int h = nn / DH, d = nn - h * DH;
          kh[((size_t)((b * NH + h) * Ttok) + t) * DHP + d] = (_Float16)v;
        } else {
          int nn = n - 2 * Hd; int h = nn / DH, d = nn - h * DH;
          vt[((size_t)((b * NH + h) * DH) + d) * Ttok + t] = (_Float16)v;
        }
      }
    }
}

// ---- Attention scores S = q @ k^T (scale folded into q) -------------------
// Wave strip: 32 q-rows x 64 t-cols (8 col groups cover 512 t).
__global__ void k_scores(const _Float16* __restrict__ qh, const _Float16* __restrict__ kh,
                         float* __restrict__ S) {
  int wid = blockIdx.x * (blockDim.x >> 5) + (threadIdx.x >> 5);
  int lane = threadIdx.x & 31;
  if (wid >= Bz * NH * 16 * 8) return;
  int ng = wid & 7, mt = (wid >> 3) & 15, bh = wid >> 7;
  const _Float16* Ab = qh + (size_t)bh * Ttok * DHP + mt * 32 * DHP;
  const _Float16* Bb = kh + (size_t)bh * Ttok * DHP + ng * 64 * DHP;
  v8f acc[8] = {};
  gemm_mm<DHP, 4, 2>(acc, Ab, DHP, Bb, DHP, lane);
  int hi = (lane >> 4) & 1, l16 = lane & 15;
  float* Sb = S + (size_t)bh * Ttok * Ttok;
#pragma unroll
  for (int m = 0; m < 2; ++m)
#pragma unroll
    for (int u = 0; u < 4; ++u)
#pragma unroll
      for (int j = 0; j < 8; ++j)
        Sb[(size_t)(mt * 32 + m * 16 + j + hi * 8) * Ttok + ng * 64 + u * 16 + l16] =
            acc[m * 4 + u][j];
}

// ---- Row softmax, write f16 P ---------------------------------------------
__global__ void k_softmax(const float* __restrict__ S, _Float16* __restrict__ P) {
  __shared__ float red[256];
  size_t base = (size_t)blockIdx.x * Ttok;
  int tid = threadIdx.x;
  float v0 = S[base + tid], v1 = S[base + 256 + tid];
  red[tid] = fmaxf(v0, v1);
  __syncthreads();
#pragma unroll
  for (int st = 128; st > 0; st >>= 1) {
    if (tid < st) red[tid] = fmaxf(red[tid], red[tid + st]);
    __syncthreads();
  }
  float m = red[0];
  __syncthreads();
  float e0 = __expf(v0 - m), e1 = __expf(v1 - m);
  red[tid] = e0 + e1;
  __syncthreads();
#pragma unroll
  for (int st = 128; st > 0; st >>= 1) {
    if (tid < st) red[tid] += red[tid + st];
    __syncthreads();
  }
  float inv = 1.f / red[0];
  P[base + tid] = (_Float16)(e0 * inv);
  P[base + 256 + tid] = (_Float16)(e1 * inv);
}

// ---- O = P @ V  (B = V^T, K=512); 32 rows x all 3 dh tiles per wave -------
__global__ void k_attnout(const _Float16* __restrict__ P, const _Float16* __restrict__ vt,
                          _Float16* __restrict__ oh) {
  int wid = blockIdx.x * (blockDim.x >> 5) + (threadIdx.x >> 5);
  int lane = threadIdx.x & 31;
  if (wid >= Bz * NH * 16) return;
  int mt = wid & 15, bh = wid >> 4;
  const _Float16* Ab = P + (size_t)bh * Ttok * Ttok + (size_t)mt * 32 * Ttok;
  const _Float16* Bb = vt + (size_t)bh * DH * Ttok;
  v8f acc[6] = {};
  gemm_mm<Ttok, 3, 2>(acc, Ab, Ttok, Bb, Ttok, lane);
  int hi = (lane >> 4) & 1, l16 = lane & 15;
  int b = bh / NH, h = bh % NH;
#pragma unroll
  for (int m = 0; m < 2; ++m)
#pragma unroll
    for (int u = 0; u < 3; ++u)
#pragma unroll
      for (int j = 0; j < 8; ++j) {
        int t = mt * 32 + m * 16 + j + hi * 8;
        oh[((size_t)(b * Ttok) + t) * Hd + h * DH + u * 16 + l16] = (_Float16)acc[m * 3 + u][j];
      }
}

// ---- GEMM 192->192 with residual add into x32 (out-proj & pointwise conv) -
__global__ void k_gemm_res(const _Float16* __restrict__ A16, const _Float16* __restrict__ W,
                           const float* __restrict__ bias, float* __restrict__ x32) {
  int wid = blockIdx.x * (blockDim.x >> 5) + (threadIdx.x >> 5);
  int lane = threadIdx.x & 31;
  if (wid >= (Bz * Ttok / 32) * 3) return;
  int ng = wid % 3, mt = wid / 3, row0 = mt * 32, n0 = ng * 64;
  v8f acc[8] = {};
  gemm_mm<Hd, 4, 2>(acc, A16 + (size_t)row0 * Hd, Hd, W + (size_t)n0 * Hd, Hd, lane);
  int hi = (lane >> 4) & 1, l16 = lane & 15;
#pragma unroll
  for (int m = 0; m < 2; ++m)
#pragma unroll
    for (int u = 0; u < 4; ++u) {
      int n = n0 + u * 16 + l16;
      float bb = bias[n];
#pragma unroll
      for (int j = 0; j < 8; ++j) {
        size_t idx = (size_t)(row0 + m * 16 + j + hi * 8) * Hd + n;
        x32[idx] = x32[idx] + acc[m * 4 + u][j] + bb;
      }
    }
}

// ---- Depthwise conv k=5 + SiLU --------------------------------------------
__global__ void k_dw(const float* __restrict__ x32, const float* __restrict__ w,
                     const float* __restrict__ bias, _Float16* __restrict__ outh) {
  int i = blockIdx.x * blockDim.x + threadIdx.x;
  if (i >= Bz * Ttok * Hd) return;
  int c = i % Hd, t = (i / Hd) % Ttok, b = i / (Hd * Ttok);
  float s = bias[c];
#pragma unroll
  for (int k = 0; k < 5; ++k) {
    int tt = t + k - 2;
    if (tt >= 0 && tt < Ttok) s += w[c * 5 + k] * x32[((size_t)(b * Ttok + tt)) * Hd + c];
  }
  s = s / (1.f + __expf(-s));
  outh[i] = (_Float16)s;
}

// ---- Duration cumsum -------------------------------------------------------
__global__ void k_cumsum(const int* __restrict__ dur, int* __restrict__ cum) {
  int b = threadIdx.x;
  if (b >= Bz) return;
  int run = 0;
  for (int t = 0; t < Ttok; ++t) { run += dur[b * Ttok + t]; cum[b * Ttok + t] = run; }
}

// ---- Length regulator: gather + mask + sinusoid PE, write f16 -------------
__global__ void k_lenreg(const float* __restrict__ x32, const int* __restrict__ cum,
                         _Float16* __restrict__ xeh) {
  int i = blockIdx.x * blockDim.x + threadIdx.x;
  if (i >= Bz * TFr * Hd) return;
  int c = i % Hd, t = (i / Hd) % TFr, b = i / (Hd * TFr);
  const int* cr = cum + b * Ttok;
  int lo = 0, hi2 = Ttok;
  while (lo < hi2) { int mid = (lo + hi2) >> 1; if (cr[mid] > t) hi2 = mid; else lo = mid + 1; }
  int tok = lo < (Ttok - 1) ? lo : (Ttok - 1);
  float mask = (t < cr[Ttok - 1]) ? 1.f : 0.f;
  int j = c >> 1;
  float f = __expf(-9.210340371976184f * (float)(2 * j) / (float)Hd);
  float ang = (float)t * f;
  float pe = (c & 1) ? cosf(ang) : sinf(ang);
  float v = x32[((size_t)(b * Ttok + tok)) * Hd + c] * mask + pe;
  xeh[i] = (_Float16)v;
}

// ---- Decoder input 1x1 conv 192->256: write y32 + padded f16 --------------
__global__ void k_decin(const _Float16* __restrict__ A16, const _Float16* __restrict__ W,
                        const float* __restrict__ bias, float* __restrict__ y32,
                        _Float16* __restrict__ yh) {
  int wid = blockIdx.x * (blockDim.x >> 5) + (threadIdx.x >> 5);
  int lane = threadIdx.x & 31;
  if (wid >= (Bz * TFr / 32) * 4) return;
  int ng = wid & 3, mt = wid >> 2, row0 = mt * 32, n0 = ng * 64;
  v8f acc[8] = {};
  gemm_mm<Hd, 4, 2>(acc, A16 + (size_t)row0 * Hd, Hd, W + (size_t)n0 * Hd, Hd, lane);
  int hi = (lane >> 4) & 1, l16 = lane & 15;
#pragma unroll
  for (int m = 0; m < 2; ++m)
#pragma unroll
    for (int u = 0; u < 4; ++u) {
      int n = n0 + u * 16 + l16;
      float bb = bias[n];
#pragma unroll
      for (int j = 0; j < 8; ++j) {
        int r = row0 + m * 16 + j + hi * 8;
        float v = acc[m * 4 + u][j] + bb;
        y32[(size_t)r * DC + n] = v;
        int b = r / TFr, t = r % TFr;
        yh[((size_t)(b * TPAD) + PADT + t) * DC + n] = (_Float16)v;
      }
    }
}

// ---- Dilated conv k=3 (3 shifted WMMA GEMM strips), f32 out ---------------
__global__ void k_conv(const _Float16* __restrict__ X, const _Float16* __restrict__ W3,
                       const float* __restrict__ bias, float* __restrict__ out32, int dil) {
  int wid = blockIdx.x * (blockDim.x >> 5) + (threadIdx.x >> 5);
  int lane = threadIdx.x & 31;
  if (wid >= Bz * (TFr / 32) * 4) return;
  int ng = wid & 3, mt = wid >> 2;
  int b = mt / (TFr / 32), t0 = (mt % (TFr / 32)) * 32, n0 = ng * 64;
  v8f acc[8] = {};
#pragma unroll
  for (int s = 0; s < 3; ++s) {
    const _Float16* A = X + ((size_t)(b * TPAD) + PADT + t0 + (s - 1) * dil) * DC;
    gemm_mm<DC, 4, 2>(acc, A, DC, W3 + (size_t)s * DC * DC + (size_t)n0 * DC, DC, lane);
  }
  int hi = (lane >> 4) & 1, l16 = lane & 15;
#pragma unroll
  for (int m = 0; m < 2; ++m)
#pragma unroll
    for (int u = 0; u < 4; ++u) {
      int n = n0 + u * 16 + l16;
      float bb = bias[n];
#pragma unroll
      for (int j = 0; j < 8; ++j)
        out32[((size_t)(b * TFr) + t0 + m * 16 + j + hi * 8) * DC + n] = acc[m * 4 + u][j] + bb;
    }
}

// ---- GroupNorm(8) + SiLU -> padded f16 ------------------------------------
__global__ void k_gn_silu(const float* __restrict__ t32, const float* __restrict__ gam,
                          const float* __restrict__ bet, _Float16* __restrict__ outh) {
  __shared__ float s1[256], s2[256];
  int b = blockIdx.x >> 3, g = blockIdx.x & 7, tid = threadIdx.x;
  float sum = 0.f, sq = 0.f;
  for (int i = tid; i < TFr * 32; i += 256) {
    int t = i >> 5, c = i & 31;
    float v = t32[((size_t)(b * TFr + t)) * DC + g * 32 + c];
    sum += v; sq += v * v;
  }
  s1[tid] = sum; s2[tid] = sq;
  __syncthreads();
#pragma unroll
  for (int st = 128; st > 0; st >>= 1) {
    if (tid < st) { s1[tid] += s1[tid + st]; s2[tid] += s2[tid + st]; }
    __syncthreads();
  }
  float mean = s1[0] * (1.f / (TFr * 32));
  float var = s2[0] * (1.f / (TFr * 32)) - mean * mean;
  float rs = rsqrtf(var + 1e-5f);
  for (int i = tid; i < TFr * 32; i += 256) {
    int t = i >> 5, c = i & 31, ch = g * 32 + c;
    float v = t32[((size_t)(b * TFr + t)) * DC + ch];
    float o = (v - mean) * rs * gam[ch] + bet[ch];
    o = o / (1.f + __expf(-o));
    outh[((size_t)(b * TPAD) + PADT + t) * DC + ch] = (_Float16)o;
  }
}

// ---- GroupNorm(8) + residual -> y32 (f32) and padded f16 ------------------
__global__ void k_gn_res(const float* __restrict__ t32, const float* __restrict__ gam,
                         const float* __restrict__ bet, float* __restrict__ y32,
                         _Float16* __restrict__ yh) {
  __shared__ float s1[256], s2[256];
  int b = blockIdx.x >> 3, g = blockIdx.x & 7, tid = threadIdx.x;
  float sum = 0.f, sq = 0.f;
  for (int i = tid; i < TFr * 32; i += 256) {
    int t = i >> 5, c = i & 31;
    float v = t32[((size_t)(b * TFr + t)) * DC + g * 32 + c];
    sum += v; sq += v * v;
  }
  s1[tid] = sum; s2[tid] = sq;
  __syncthreads();
#pragma unroll
  for (int st = 128; st > 0; st >>= 1) {
    if (tid < st) { s1[tid] += s1[tid + st]; s2[tid] += s2[tid + st]; }
    __syncthreads();
  }
  float mean = s1[0] * (1.f / (TFr * 32));
  float var = s2[0] * (1.f / (TFr * 32)) - mean * mean;
  float rs = rsqrtf(var + 1e-5f);
  for (int i = tid; i < TFr * 32; i += 256) {
    int t = i >> 5, c = i & 31, ch = g * 32 + c;
    size_t idx = ((size_t)(b * TFr + t)) * DC + ch;
    float v = t32[idx];
    float o = (v - mean) * rs * gam[ch] + bet[ch] + y32[idx];
    y32[idx] = o;
    yh[((size_t)(b * TPAD) + PADT + t) * DC + ch] = (_Float16)o;
  }
}

// ---- Mel projection 256->80 -> d_out (f32); 32 rows x all 5 N tiles -------
__global__ void k_outconv(const _Float16* __restrict__ yh, const _Float16* __restrict__ W,
                          const float* __restrict__ bias, float* __restrict__ out) {
  int wid = blockIdx.x * (blockDim.x >> 5) + (threadIdx.x >> 5);
  int lane = threadIdx.x & 31;
  if (wid >= Bz * TFr / 32) return;
  int b = wid / (TFr / 32), t0 = (wid % (TFr / 32)) * 32;
  const _Float16* A = yh + ((size_t)(b * TPAD) + PADT + t0) * DC;
  v8f acc[10] = {};
  gemm_mm<DC, 5, 2>(acc, A, DC, W, DC, lane);
  int hi = (lane >> 4) & 1, l16 = lane & 15;
#pragma unroll
  for (int m = 0; m < 2; ++m)
#pragma unroll
    for (int u = 0; u < 5; ++u) {
      int n = u * 16 + l16;
      float bb = bias[n];
#pragma unroll
      for (int j = 0; j < 8; ++j)
        out[((size_t)(b * TFr) + t0 + m * 16 + j + hi * 8) * MELC + n] = acc[m * 5 + u][j] + bb;
    }
}

// ---------------------------------------------------------------------------
extern "C" void kernel_launch(void* const* d_in, const int* in_sizes, int n_in,
                              void* d_out, int out_size, void* d_ws, size_t ws_size,
                              hipStream_t stream) {
  (void)n_in; (void)out_size; (void)ws_size;
  // Top-level inputs disambiguated by element count.
  const float* bert = nullptr; const int* eid = nullptr; const int* dur = nullptr;
  for (int i = 0; i < 3; ++i) {
    if (in_sizes[i] == Bz * Ttok * DIN) bert = (const float*)d_in[i];
    else if (in_sizes[i] == Bz) eid = (const int*)d_in[i];
    else dur = (const int*)d_in[i];
  }
  // Param leaves: JAX pytree order (dict keys alphabetical at every level).
  const int PB = 3;
  auto F = [&](int i) { return (const float*)d_in[PB + i]; };
  const float *c1_b[4], *c1_w[4], *c2_b[4], *c2_w[4], *g1_b[4], *g1_g[4], *g2_b[4], *g2_g[4];
  for (int j = 0; j < 4; ++j) {
    c1_b[j] = F(j * 8 + 0); c1_w[j] = F(j * 8 + 1);
    c2_b[j] = F(j * 8 + 2); c2_w[j] = F(j * 8 + 3);
    g1_b[j] = F(j * 8 + 4); g1_g[j] = F(j * 8 + 5);
    g2_b[j] = F(j * 8 + 6); g2_g[j] = F(j * 8 + 7);
  }
  const float* dec_in_b = F(32); const float* dec_in_w = F(33);
  const float* dec_out_b = F(34); const float* dec_out_w = F(35);
  const float* emo = F(36);
  const float *dw_b[2], *dw_w[2], *ln_b[2], *ln_g[2], *out_b[2], *out_w[2],
              *pw_b[2], *pw_w[2], *qkv_b[2], *qkv_w[2];
  for (int l = 0; l < 2; ++l) {
    int bs = 37 + l * 10;
    dw_b[l] = F(bs + 0); dw_w[l] = F(bs + 1); ln_b[l] = F(bs + 2); ln_g[l] = F(bs + 3);
    out_b[l] = F(bs + 4); out_w[l] = F(bs + 5); pw_b[l] = F(bs + 6); pw_w[l] = F(bs + 7);
    qkv_b[l] = F(bs + 8); qkv_w[l] = F(bs + 9);
  }
  const float* in_b = F(57); const float* in_w = F(58);

  // Workspace arena (256B aligned allocations).
  char* basep = (char*)d_ws; size_t off = 0;
  auto alloc = [&](size_t bytes) -> void* {
    void* p = basep + off; off += (bytes + 255) & ~(size_t)255; return p;
  };
  _Float16* w_in_h = (_Float16*)alloc(Hd * DIN * 2);
  _Float16* w_qkv_h[2]; for (int l = 0; l < 2; ++l) w_qkv_h[l] = (_Float16*)alloc(3 * Hd * Hd * 2);
  _Float16* w_out_h[2]; for (int l = 0; l < 2; ++l) w_out_h[l] = (_Float16*)alloc(Hd * Hd * 2);
  _Float16* w_pw_h[2];  for (int l = 0; l < 2; ++l) w_pw_h[l] = (_Float16*)alloc(Hd * Hd * 2);
  _Float16* w_decin_h = (_Float16*)alloc(DC * Hd * 2);
  _Float16* w_decout_h = (_Float16*)alloc(MELC * DC * 2);
  _Float16* w_c1[4]; _Float16* w_c2[4];
  for (int j = 0; j < 4; ++j) { w_c1[j] = (_Float16*)alloc(3 * DC * DC * 2); w_c2[j] = (_Float16*)alloc(3 * DC * DC * 2); }
  float* x32 = (float*)alloc((size_t)Bz * Ttok * Hd * 4);
  _Float16* berth = (_Float16*)alloc((size_t)Bz * Ttok * DIN * 2);
  _Float16* nh  = (_Float16*)alloc((size_t)Bz * Ttok * Hd * 2);
  _Float16* oh  = (_Float16*)alloc((size_t)Bz * Ttok * Hd * 2);
  _Float16* dwh = (_Float16*)alloc((size_t)Bz * Ttok * Hd * 2);
  _Float16* vt  = (_Float16*)alloc((size_t)Bz * NH * DH * Ttok * 2);
  _Float16* xeh = (_Float16*)alloc((size_t)Bz * TFr * Hd * 2);
  int* cum = (int*)alloc((size_t)Bz * Ttok * 4);
  // Contiguous zero region: qh, kh (dh pad lanes), yh, hh (time halos).
  const size_t QSZ = (size_t)Bz * NH * Ttok * DHP * 2;        // 8,388,608 B
  const size_t YSZ = (size_t)Bz * TPAD * DC * 2;              // 42,205,184 B
  _Float16* qh = (_Float16*)alloc(QSZ);
  _Float16* kh = (_Float16*)alloc(QSZ);
  _Float16* yh = (_Float16*)alloc(YSZ);
  _Float16* hh = (_Float16*)alloc(YSZ);
  // Big arena: attention S(f32)+P(f16) reused as decoder y32+t32 (phases disjoint).
  const size_t SSZ = (size_t)Bz * NH * Ttok * Ttok * 4;       // 134 MB
  const size_t PSZ = SSZ / 2;                                  // 67 MB
  const size_t Y32SZ = (size_t)Bz * TFr * DC * 4;              // 84 MB
  char* big = (char*)alloc((SSZ + PSZ > 2 * Y32SZ) ? (SSZ + PSZ) : (2 * Y32SZ));
  float* S32 = (float*)big;
  _Float16* ph = (_Float16*)(big + SSZ);
  float* y32 = (float*)big;
  float* t32 = (float*)(big + Y32SZ);

  // 0) zero pads/halos (qh|kh|yh|hh are contiguous, sizes multiple of 256)
  size_t zbytes = 2 * QSZ + 2 * YSZ;
  k_zero16<<<4096, 256, 0, stream>>>((uint4*)qh, zbytes / 16);

  // 1) weight + input conversions to f16
  auto cv = [&](const float* s, _Float16* d, int n) {
    k_f32h<<<(n + 255) / 256, 256, 0, stream>>>(s, d, n);
  };
  cv(in_w, w_in_h, Hd * DIN);
  for (int l = 0; l < 2; ++l) {
    cv(qkv_w[l], w_qkv_h[l], 3 * Hd * Hd);
    cv(out_w[l], w_out_h[l], Hd * Hd);
    cv(pw_w[l], w_pw_h[l], Hd * Hd);
  }
  cv(dec_in_w, w_decin_h, DC * Hd);
  cv(dec_out_w, w_decout_h, MELC * DC);
  cv(bert, berth, Bz * Ttok * DIN);
  for (int j = 0; j < 4; ++j) {
    k_convw<<<(3 * DC * DC + 255) / 256, 256, 0, stream>>>(c1_w[j], w_c1[j]);
    k_convw<<<(3 * DC * DC + 255) / 256, 256, 0, stream>>>(c2_w[j], w_c2[j]);
  }

  // 2) input projection + emotion embedding  (1536 waves / 8 per block)
  k_inproj<<<192, 256, 0, stream>>>(berth, w_in_h, in_b, emo, eid, x32);

  // 3) encoder layers
  for (int l = 0; l < 2; ++l) {
    k_ln<<<Bz * Ttok, 64, 0, stream>>>(x32, ln_g[l], ln_b[l], nh);
    k_qkv<<<576, 256, 0, stream>>>(nh, w_qkv_h[l], qkv_b[l], qh, kh, vt);        // 4608 waves
    k_scores<<<2048, 256, 0, stream>>>(qh, kh, S32);                              // 16384 waves
    k_softmax<<<Bz * NH * Ttok, 256, 0, stream>>>(S32, ph);
    k_attnout<<<256, 256, 0, stream>>>(ph, vt, oh);                               // 2048 waves
    k_gemm_res<<<192, 256, 0, stream>>>(oh, w_out_h[l], out_b[l], x32);
    k_dw<<<(Bz * Ttok * Hd + 255) / 256, 256, 0, stream>>>(x32, dw_w[l], dw_b[l], dwh);
    k_gemm_res<<<192, 256, 0, stream>>>(dwh, w_pw_h[l], pw_b[l], x32);
  }

  // 4) length regulator
  k_cumsum<<<1, 32, 0, stream>>>(dur, cum);
  k_lenreg<<<(Bz * TFr * Hd + 255) / 256, 256, 0, stream>>>(x32, cum, xeh);

  // 5) decoder
  k_decin<<<1280, 256, 0, stream>>>(xeh, w_decin_h, dec_in_b, y32, yh);           // 10240 waves
  int dil = 1;
  for (int j = 0; j < 4; ++j) {
    k_conv<<<1280, 256, 0, stream>>>(yh, w_c1[j], c1_b[j], t32, dil);             // 10240 waves
    k_gn_silu<<<Bz * 8, 256, 0, stream>>>(t32, g1_g[j], g1_b[j], hh);
    k_conv<<<1280, 256, 0, stream>>>(hh, w_c2[j], c2_b[j], t32, dil);
    k_gn_res<<<Bz * 8, 256, 0, stream>>>(t32, g2_g[j], g2_b[j], y32, yh);
    dil <<= 1;
  }
  k_outconv<<<320, 256, 0, stream>>>(yh, w_decout_h, dec_out_b, (float*)d_out);
}
